// SelfCausalAttention_52218212385323
// MI455X (gfx1250) — matrix operand order repaired and checked by reading, not compiled, not run on hardware
//
#include <hip/hip_runtime.h>
#include <hip/hip_bf16.h>
#include <math.h>

typedef __attribute__((ext_vector_type(16))) __bf16 v16bf;
typedef __attribute__((ext_vector_type(8)))  float  v8f;

#define BT   2048
#define HD   64
#define NH   16
#define CEMB 1024

// ---- CDNA5 async global->LDS copy (guarded; falls back to manual staging) --
#if defined(__gfx1250__) &&                                             \
    __has_builtin(__builtin_amdgcn_global_load_async_to_lds_b128) &&     \
    __has_builtin(__builtin_amdgcn_s_wait_asynccnt)
#define HAVE_ASYNC 1
typedef __attribute__((__vector_size__(4 * sizeof(int)))) int gvec4i;
typedef __attribute__((address_space(1))) gvec4i* as1_v4i;
typedef __attribute__((address_space(3))) gvec4i* as3_v4i;
#else
#define HAVE_ASYNC 0
#endif

static __device__ __forceinline__ void cp16_to_lds(const __bf16* g, __bf16* l) {
#if HAVE_ASYNC
  // 16 bytes per lane, DMA'd straight into LDS (ASYNCcnt-tracked)
  __builtin_amdgcn_global_load_async_to_lds_b128(
      (as1_v4i)(void*)g, (as3_v4i)(void*)l, 0, 0);
#else
  *(uint4*)l = *(const uint4*)g;
#endif
}
static __device__ __forceinline__ void async_wait_all() {
#if HAVE_ASYNC
  __builtin_amdgcn_s_wait_asynccnt(0);
#endif
}

// ---- fragment helpers ------------------------------------------------------
// A-matrix 16x32 bf16 layout: lane half kg holds K = kg*16 + j (contiguous 16)
static __device__ __forceinline__ v16bf frag_a(const __bf16* p) {
  v16bf r;
#pragma unroll
  for (int i = 0; i < 16; ++i) r[i] = p[i];
  return r;
}
// B-matrix 32x16 bf16 layout: lane half kg holds K = kg*16 + j (contiguous 16)
static __device__ __forceinline__ v16bf frag_b(const __bf16* p) {
  v16bf r;
#pragma unroll
  for (int i = 0; i < 16; ++i) r[i] = p[i];
  return r;
}
static __device__ __forceinline__ v8f wmma_bf16(v16bf a, v16bf b, v8f c) {
  return __builtin_amdgcn_wmma_f32_16x16x32_bf16(false, a, false, b,
                                                 (short)0, c, false, false);
}
static __device__ __forceinline__ v8f zero8() {
  v8f z;
#pragma unroll
  for (int i = 0; i < 8; ++i) z[i] = 0.f;
  return z;
}

// ---- staging helpers -------------------------------------------------------
// A tile: 128x32 bf16 row-major, 16 elems/thread (2x b128 async)
static __device__ __forceinline__ void stage_A(const __bf16* A, __bf16* As,
                                               int row0, int k0, int Kdim,
                                               int tid) {
  int r = tid >> 1, c = (tid & 1) * 16;
  const __bf16* g = A + (size_t)(row0 + r) * Kdim + k0 + c;
  __bf16* l = As + r * 32 + c;
  cp16_to_lds(g, l);
  cp16_to_lds(g + 8, l + 8);
}
// B tile: 32x64 -> transposed Bt[col][k], 8 elems/thread (manual: transpose)
static __device__ __forceinline__ void stage_Bt(const __bf16* B, __bf16* Bt,
                                                int col0, int k0, int N,
                                                int tid) {
  int r = tid >> 3, c = (tid & 7) * 8;
  union { uint4 u; __bf16 h[8]; } t;
  t.u = *(const uint4*)(B + (size_t)(k0 + r) * N + col0 + c);
#pragma unroll
  for (int j = 0; j < 8; ++j) Bt[(c + j) * 32 + r] = t.h[j];
}

// ---- fp32 -> bf16 conversion ----------------------------------------------
__global__ void cvt_f32_bf16(const float* __restrict__ src,
                             __bf16* __restrict__ dst, int n4) {
  int i = blockIdx.x * blockDim.x + threadIdx.x;
  int stride = gridDim.x * blockDim.x;
  for (; i < n4; i += stride) {
    float4 f = ((const float4*)src)[i];
    dst[4 * i + 0] = (__bf16)f.x;
    dst[4 * i + 1] = (__bf16)f.y;
    dst[4 * i + 2] = (__bf16)f.z;
    dst[4 * i + 3] = (__bf16)f.w;
  }
}

// ---- QKV GEMM: [8192,1024]x[1024,3072]+bias -> Q/K/V [B,H,T,d] bf16 -------
__global__ __launch_bounds__(256) void qkv_gemm(
    const __bf16* __restrict__ A, const __bf16* __restrict__ B,
    const float* __restrict__ bias,
    __bf16* __restrict__ Qo, __bf16* __restrict__ Ko, __bf16* __restrict__ Vo) {
  const int Kdim = 1024, N = 3072;
  __shared__ __align__(16) __bf16 As[2][128 * 32];
  __shared__ __align__(16) __bf16 Bt[2][64 * 32];
  const int tid = threadIdx.x;
  const int lane = tid & 31, w = tid >> 5;
  const int wm = w >> 1, wn = w & 1;
  const int row0 = blockIdx.y * 128;
  const int col0 = blockIdx.x * 64;
  const int kg = lane >> 4, ln = lane & 15;

  v8f acc[2][2];
#pragma unroll
  for (int mi = 0; mi < 2; ++mi)
#pragma unroll
    for (int ni = 0; ni < 2; ++ni) acc[mi][ni] = zero8();

  stage_A(A, As[0], row0, 0, Kdim, tid);
  stage_Bt(B, Bt[0], col0, 0, N, tid);

  const int nk = Kdim / 32;
  for (int kb = 0; kb < nk; ++kb) {
    const int cur = kb & 1, nxt = cur ^ 1;
    async_wait_all();
    __syncthreads();
    if (kb + 1 < nk) {  // issue next tile's DMA before computing current
      stage_A(A, As[nxt], row0, (kb + 1) * 32, Kdim, tid);
      stage_Bt(B, Bt[nxt], col0, (kb + 1) * 32, N, tid);
    }
    const __bf16* Ac = As[cur];
    const __bf16* Bc = Bt[cur];
    v16bf a0 = frag_a(Ac + (wm * 32 + ln) * 32 + kg * 16);
    v16bf a1 = frag_a(Ac + (wm * 32 + 16 + ln) * 32 + kg * 16);
    v16bf b0 = frag_b(Bc + (wn * 32 + ln) * 32 + kg * 16);
    v16bf b1 = frag_b(Bc + (wn * 32 + 16 + ln) * 32 + kg * 16);
    acc[0][0] = wmma_bf16(a0, b0, acc[0][0]);
    acc[0][1] = wmma_bf16(a0, b1, acc[0][1]);
    acc[1][0] = wmma_bf16(a1, b0, acc[1][0]);
    acc[1][1] = wmma_bf16(a1, b1, acc[1][1]);
  }
  // epilogue: bias + scatter to Q/K/V in [B,H,T,d]
#pragma unroll
  for (int mi = 0; mi < 2; ++mi)
#pragma unroll
    for (int ni = 0; ni < 2; ++ni)
#pragma unroll
      for (int v = 0; v < 8; ++v) {
        int m = row0 + wm * 32 + mi * 16 + v + 8 * kg;
        int n = col0 + wn * 32 + ni * 16 + ln;
        float val = acc[mi][ni][v] + bias[n];
        int b = m >> 11, t = m & (BT - 1);
        __bf16* dst; int nn;
        if (n < 1024)      { dst = Qo; nn = n; }
        else if (n < 2048) { dst = Ko; nn = n - 1024; }
        else               { dst = Vo; nn = n - 2048; }
        int h = nn >> 6, dd = nn & 63;
        dst[(((size_t)(b * NH + h)) * BT + t) * HD + dd] = (__bf16)val;
      }
}

// ---- Flash attention, causal, scale applied AFTER softmax (source quirk) ---
__global__ __launch_bounds__(256) void attn_kernel(
    const __bf16* __restrict__ Q, const __bf16* __restrict__ K,
    const __bf16* __restrict__ V, __bf16* __restrict__ O) {
  __shared__ __align__(16) __bf16 Ks[2][32 * 64];    // [key][d]
  __shared__ __align__(16) __bf16 Vt[2][64 * 32];    // [d][key]
  __shared__ __align__(16) __bf16 Pb[8][16 * 32];    // per-wave P staging
  const int tid = threadIdx.x, lane = tid & 31, w = tid >> 5;
  const int kg = lane >> 4, ln = lane & 15;
  const int q0 = blockIdx.x * 128;
  const int bh = blockIdx.y;  // b*16 + h
  const __bf16* Qp = Q + (size_t)bh * BT * HD;
  const __bf16* Kp = K + (size_t)bh * BT * HD;
  const __bf16* Vp = V + (size_t)bh * BT * HD;
  const int qw = q0 + w * 16;  // this wave's first query row

  v16bf aq0 = frag_a(Qp + (size_t)(qw + ln) * HD + 0  + kg * 16);
  v16bf aq1 = frag_a(Qp + (size_t)(qw + ln) * HD + 32 + kg * 16);

  v8f o[4];
#pragma unroll
  for (int dt = 0; dt < 4; ++dt) o[dt] = zero8();
  float m_[8], l_[8];
#pragma unroll
  for (int v = 0; v < 8; ++v) { m_[v] = -1e30f; l_[v] = 0.f; }

  // staging indices (K async row-major, V manual transpose)
  const int sr = tid >> 3, sc = (tid & 7) * 8;

  // prologue: stage key-block 0
  cp16_to_lds(Kp + (size_t)sr * HD + sc, Ks[0] + sr * 64 + sc);
  {
    union { uint4 u; __bf16 h[8]; } t;
    t.u = *(const uint4*)(Vp + (size_t)sr * HD + sc);
#pragma unroll
    for (int j = 0; j < 8; ++j) Vt[0][(sc + j) * 32 + sr] = t.h[j];
  }

  const int nkb = (q0 + 128) / 32;
  for (int kb = 0; kb < nkb; ++kb) {
    const int ks = kb * 32;
    const int cur = kb & 1, nxt = cur ^ 1;
    async_wait_all();
    __syncthreads();
    if (kb + 1 < nkb) {  // DMA next K/V tile while computing this one
      const int ks2 = ks + 32;
      cp16_to_lds(Kp + (size_t)(ks2 + sr) * HD + sc, Ks[nxt] + sr * 64 + sc);
      union { uint4 u; __bf16 h[8]; } t;
      t.u = *(const uint4*)(Vp + (size_t)(ks2 + sr) * HD + sc);
#pragma unroll
      for (int j = 0; j < 8; ++j) Vt[nxt][(sc + j) * 32 + sr] = t.h[j];
    }
    const __bf16* Kc = Ks[cur];
    const __bf16* Vc = Vt[cur];

    // S = Q @ K^T : two 16-key sub-tiles, contraction d=64 (2 WMMA steps)
    v8f s0 = zero8(), s1 = zero8();
    s0 = wmma_bf16(aq0, frag_b(Kc + (0 * 16 + ln) * 64 + 0  + kg * 16), s0);
    s0 = wmma_bf16(aq1, frag_b(Kc + (0 * 16 + ln) * 64 + 32 + kg * 16), s0);
    s1 = wmma_bf16(aq0, frag_b(Kc + (1 * 16 + ln) * 64 + 0  + kg * 16), s1);
    s1 = wmma_bf16(aq1, frag_b(Kc + (1 * 16 + ln) * 64 + 32 + kg * 16), s1);

    // online softmax in C/D layout: row = v + 8*kg, col = ln
    __bf16* P = &Pb[w][0];
#pragma unroll
    for (int v = 0; v < 8; ++v) {
      int rq = qw + v + 8 * kg;
      float a = s0[v]; if (ks + ln > rq)      a = -1e30f;
      float b = s1[v]; if (ks + 16 + ln > rq) b = -1e30f;
      float mv = fmaxf(a, b);
      mv = fmaxf(mv, __shfl_xor(mv, 1, 32));
      mv = fmaxf(mv, __shfl_xor(mv, 2, 32));
      mv = fmaxf(mv, __shfl_xor(mv, 4, 32));
      mv = fmaxf(mv, __shfl_xor(mv, 8, 32));
      float mn  = fmaxf(m_[v], mv);
      float fac = __expf(m_[v] - mn);
      float p0 = __expf(a - mn);
      float p1 = __expf(b - mn);
      float sum = p0 + p1;
      sum += __shfl_xor(sum, 1, 32);
      sum += __shfl_xor(sum, 2, 32);
      sum += __shfl_xor(sum, 4, 32);
      sum += __shfl_xor(sum, 8, 32);
      l_[v] = l_[v] * fac + sum;
      m_[v] = mn;
#pragma unroll
      for (int dt = 0; dt < 4; ++dt) o[dt][v] *= fac;
      P[(v + 8 * kg) * 32 + ln]      = (__bf16)p0;
      P[(v + 8 * kg) * 32 + 16 + ln] = (__bf16)p1;  // relayout C/D -> A
    }
    // O += P @ V (K=32 contraction); same-wave LDS ops stay ordered
    v16bf ap = frag_a(P + ln * 32 + kg * 16);
#pragma unroll
    for (int dt = 0; dt < 4; ++dt)
      o[dt] = wmma_bf16(ap, frag_b(Vc + (dt * 16 + ln) * 32 + kg * 16), o[dt]);
  }

  // epilogue: O[b, t, h*64+d] = acc/l * C^-0.5 (post-softmax scale, folded)
  const int b = bh >> 4, h = bh & 15;
  const float scale = 0.03125f;  // 1024^-0.5
#pragma unroll
  for (int dt = 0; dt < 4; ++dt)
#pragma unroll
    for (int v = 0; v < 8; ++v) {
      int t = qw + v + 8 * kg;
      int d = dt * 16 + ln;
      float val = o[dt][v] / l_[v] * scale;
      O[((size_t)(b * BT + t)) * CEMB + h * HD + d] = (__bf16)val;
    }
}

// ---- Output projection: [8192,1024]x[1024,1024]+bias -> fp32 out ----------
__global__ __launch_bounds__(256) void proj_gemm(
    const __bf16* __restrict__ A, const __bf16* __restrict__ B,
    const float* __restrict__ bias, float* __restrict__ Cout) {
  const int Kdim = 1024, N = 1024;
  __shared__ __align__(16) __bf16 As[2][128 * 32];
  __shared__ __align__(16) __bf16 Bt[2][64 * 32];
  const int tid = threadIdx.x;
  const int lane = tid & 31, w = tid >> 5;
  const int wm = w >> 1, wn = w & 1;
  const int row0 = blockIdx.y * 128;
  const int col0 = blockIdx.x * 64;
  const int kg = lane >> 4, ln = lane & 15;

  v8f acc[2][2];
#pragma unroll
  for (int mi = 0; mi < 2; ++mi)
#pragma unroll
    for (int ni = 0; ni < 2; ++ni) acc[mi][ni] = zero8();

  stage_A(A, As[0], row0, 0, Kdim, tid);
  stage_Bt(B, Bt[0], col0, 0, N, tid);

  const int nk = Kdim / 32;
  for (int kb = 0; kb < nk; ++kb) {
    const int cur = kb & 1, nxt = cur ^ 1;
    async_wait_all();
    __syncthreads();
    if (kb + 1 < nk) {
      stage_A(A, As[nxt], row0, (kb + 1) * 32, Kdim, tid);
      stage_Bt(B, Bt[nxt], col0, (kb + 1) * 32, N, tid);
    }
    const __bf16* Ac = As[cur];
    const __bf16* Bc = Bt[cur];
    v16bf a0 = frag_a(Ac + (wm * 32 + ln) * 32 + kg * 16);
    v16bf a1 = frag_a(Ac + (wm * 32 + 16 + ln) * 32 + kg * 16);
    v16bf b0 = frag_b(Bc + (wn * 32 + ln) * 32 + kg * 16);
    v16bf b1 = frag_b(Bc + (wn * 32 + 16 + ln) * 32 + kg * 16);
    acc[0][0] = wmma_bf16(a0, b0, acc[0][0]);
    acc[0][1] = wmma_bf16(a0, b1, acc[0][1]);
    acc[1][0] = wmma_bf16(a1, b0, acc[1][0]);
    acc[1][1] = wmma_bf16(a1, b1, acc[1][1]);
  }
#pragma unroll
  for (int mi = 0; mi < 2; ++mi)
#pragma unroll
    for (int ni = 0; ni < 2; ++ni)
#pragma unroll
      for (int v = 0; v < 8; ++v) {
        int m = row0 + wm * 32 + mi * 16 + v + 8 * kg;
        int n = col0 + wn * 32 + ni * 16 + ln;
        Cout[(size_t)m * N + n] = acc[mi][ni][v] + bias[n];
      }
}

extern "C" void kernel_launch(void* const* d_in, const int* in_sizes, int n_in,
                              void* d_out, int out_size, void* d_ws,
                              size_t ws_size, hipStream_t stream) {
  const float* x      = (const float*)d_in[0];
  const float* w_attn = (const float*)d_in[1];
  const float* b_attn = (const float*)d_in[2];
  const float* w_proj = (const float*)d_in[3];
  const float* b_proj = (const float*)d_in[4];
  float* out = (float*)d_out;

  __bf16* xb  = (__bf16*)d_ws;                      // 8192*1024
  __bf16* wab = xb  + (size_t)8192 * 1024;          // 1024*3072
  __bf16* wpb = wab + (size_t)1024 * 3072;          // 1024*1024
  __bf16* Qb  = wpb + (size_t)1024 * 1024;          // 64*2048*64
  __bf16* Kb  = Qb  + (size_t)64 * 2048 * 64;
  __bf16* Vb  = Kb  + (size_t)64 * 2048 * 64;
  __bf16* Ob  = Vb  + (size_t)64 * 2048 * 64;       // 8192*1024

  cvt_f32_bf16<<<2048, 256, 0, stream>>>(x, xb, (8192 * 1024) / 4);
  cvt_f32_bf16<<<768,  256, 0, stream>>>(w_attn, wab, (1024 * 3072) / 4);
  cvt_f32_bf16<<<256,  256, 0, stream>>>(w_proj, wpb, (1024 * 1024) / 4);

  qkv_gemm<<<dim3(3072 / 64, 8192 / 128), 256, 0, stream>>>(xb, wab, b_attn,
                                                            Qb, Kb, Vb);
  attn_kernel<<<dim3(2048 / 128, 64), 256, 0, stream>>>(Qb, Kb, Vb, Ob);
  proj_gemm<<<dim3(1024 / 64, 8192 / 128), 256, 0, stream>>>(Ob, wpb, b_proj,
                                                             out);
}
